// InterpretableMultiHead_5016521801975
// MI455X (gfx1250) — compile-verified, matrix operand order checked by reading, not run on hardware
//
#include <hip/hip_runtime.h>
#include <hip/hip_bf16.h>

// ---------------------------------------------------------------------------
// InterpretableMultiHead for MI455X (gfx1250): all GEMMs via
// v_wmma_f32_16x16x32_bf16 (wave32), flash-attention inner loop, LDS transpose
// for the P fragment, cross-wave head-mean reduction via ds_add_f32.
// Round 4: projection GEMM K-loops fully unrolled (no loop-carried fragment
// registers -> no v_mov copies, immediate-offset loads, scheduler-driven
// pipelining); attention keeps the ping-pong K double buffer.
// ---------------------------------------------------------------------------

typedef __attribute__((ext_vector_type(16))) __bf16 v16bf;
typedef __attribute__((ext_vector_type(8)))  __bf16 v8bf;
typedef __attribute__((ext_vector_type(8)))  float  v8f;
typedef __attribute__((ext_vector_type(4)))  float  v4f;

#define N_B   8
#define N_S   1024
#define N_H   16
#define N_D   1024
#define N_DH  64
#define BS    (N_B * N_S)   // 8192 token rows

// ---- helpers ---------------------------------------------------------------

__device__ __forceinline__ __bf16 f2bf(float f) {
  union { float f; unsigned u; } v; v.f = f;
  unsigned r = (v.u + 0x7FFFu + ((v.u >> 16) & 1u)) >> 16;  // RNE
  union { unsigned short s; __bf16 b; } o; o.s = (unsigned short)r;
  return o.b;
}

// Load a 16-element bf16 WMMA fragment whose two 8-element k-chunks live at
// p[0..7] and p[16..23] (the 16-bit A/B VGPR layout: lane half selects +8,
// caller already applied `half*8`). Both chunks are 16B aligned -> B128 loads.
__device__ __forceinline__ v16bf frag_ld(const __bf16* p) {
  v8bf c0 = *(const v8bf*)(p);
  v8bf c1 = *(const v8bf*)(p + 16);
  v16bf a;
#pragma unroll
  for (int i = 0; i < 8; ++i) { a[i] = c0[i]; a[8 + i] = c1[i]; }
  return a;
}

__device__ __forceinline__ v8f wmma_bf16(v16bf a, v16bf b, v8f c) {
  // (neg_a, A, neg_b, B, c_mod, C, reuse_a, reuse_b)
  return __builtin_amdgcn_wmma_f32_16x16x32_bf16(false, a, false, b,
                                                 (short)0, c, false, false);
}

// ---- elementwise conversions ----------------------------------------------

__global__ __launch_bounds__(256) void cvt_bf16_kernel(const float* __restrict__ in,
                                                       __bf16* __restrict__ out,
                                                       long n) {
  long i = (long)blockIdx.x * blockDim.x + threadIdx.x;
  if (i < n) out[i] = f2bf(in[i]);
}

// in: [batch][R][C] fp32  ->  out: [batch][C][R] bf16   (batched transpose)
__global__ __launch_bounds__(256) void btrans_bf16_kernel(const float* __restrict__ in,
                                                          __bf16* __restrict__ out,
                                                          int R, int C, long total) {
  long i = (long)blockIdx.x * blockDim.x + threadIdx.x;
  if (i >= total) return;
  long rc = (long)R * C;
  long bb = i / rc;
  long rem = i - bb * rc;
  int c = (int)(rem / R);
  int r = (int)(rem - (long)c * R);
  out[i] = f2bf(in[bb * rc + (long)r * C + c]);
}

// ---- per-head projection GEMM:  Out[b,h,s,:] = (X[b,s,:] @ W[h] + bias[h]) * scale
// X: [BS][1024] bf16, Wt: [H][64][1024] bf16 (transposed), Out: [B][H][S][64] bf16
// K loop fully unrolled: straight-line code, no loop-carried fragments, the
// scheduler hoists the B128 load clauses across WMMAs freely.
__global__ __launch_bounds__(256) void proj_qk_kernel(const __bf16* __restrict__ Xb,
                                                      const __bf16* __restrict__ Wt,
                                                      const float* __restrict__ bias,
                                                      __bf16* __restrict__ Out,
                                                      float scale) {
  int wave = threadIdx.x >> 5;
  int lane = threadIdx.x & 31;
  int lgrp = lane & 15, half = lane >> 4;
  int h = blockIdx.y;
  int rbase = (blockIdx.x * 8 + wave) * 16;      // 16 token rows per wave

  const __bf16* xr = Xb + (size_t)(rbase + lgrp) * N_D + half * 8;
  const __bf16* wr = Wt + (size_t)h * N_DH * N_D + (size_t)lgrp * N_D + half * 8;
  const size_t wstep = (size_t)16 * N_D;         // 16 weight rows per n-tile

  v8f acc[4] = {};

#pragma unroll
  for (int kc = 0; kc < N_D / 32; ++kc) {
    v16bf a = frag_ld(xr + kc * 32);
#pragma unroll
    for (int t = 0; t < 4; ++t) {
      v16bf bw = frag_ld(wr + t * wstep + kc * 32);
      acc[t] = wmma_bf16(a, bw, acc[t]);
    }
  }

  int b = rbase / N_S;
  int s0 = rbase % N_S;
  __bf16* orow = Out + ((size_t)(b * N_H + h) * N_S) * N_DH;
#pragma unroll
  for (int t = 0; t < 4; ++t) {
    float bb = bias[h * N_DH + t * 16 + lgrp];
#pragma unroll
    for (int r = 0; r < 8; ++r) {
      int s = s0 + r + 8 * half;
      orow[(size_t)s * N_DH + t * 16 + lgrp] = f2bf((acc[t][r] + bb) * scale);
    }
  }
}

// shared V projection, stored TRANSPOSED:  Vt[b][dh][s] = value[b,s,:]@Wv + bv
__global__ __launch_bounds__(256) void proj_v_kernel(const __bf16* __restrict__ Xb,
                                                     const __bf16* __restrict__ Wvt,
                                                     const float* __restrict__ bv,
                                                     __bf16* __restrict__ Vt) {
  int wave = threadIdx.x >> 5;
  int lane = threadIdx.x & 31;
  int lgrp = lane & 15, half = lane >> 4;
  int rbase = (blockIdx.x * 8 + wave) * 16;

  const __bf16* xr = Xb + (size_t)(rbase + lgrp) * N_D + half * 8;
  const __bf16* wr = Wvt + (size_t)lgrp * N_D + half * 8;
  const size_t wstep = (size_t)16 * N_D;

  v8f acc[4] = {};

#pragma unroll
  for (int kc = 0; kc < N_D / 32; ++kc) {
    v16bf a = frag_ld(xr + kc * 32);
#pragma unroll
    for (int t = 0; t < 4; ++t) {
      v16bf bw = frag_ld(wr + t * wstep + kc * 32);
      acc[t] = wmma_bf16(a, bw, acc[t]);
    }
  }

  int b = rbase / N_S;
  int s0 = rbase % N_S;
#pragma unroll
  for (int t = 0; t < 4; ++t) {
    int n = t * 16 + lgrp;
    float bb = bv[n];
#pragma unroll
    for (int r = 0; r < 8; ++r) {
      int s = s0 + r + 8 * half;
      Vt[((size_t)b * N_DH + n) * N_S + s] = f2bf(acc[t][r] + bb);
    }
  }
}

// ---- flash attention + head mean -------------------------------------------
// grid: (b, S/16); block: 512 threads = 16 waves, one head per wave.
// Q was pre-scaled by d_head^-0.5 in the projection.
__device__ __forceinline__ void load_k_frags(const __bf16* kb, int k0,
                                             int lgrp, int half, v16bf bk[4]) {
  const __bf16* kp0 = kb + (size_t)(k0 + lgrp) * N_DH + half * 8;
  const __bf16* kp1 = kb + (size_t)(k0 + 16 + lgrp) * N_DH + half * 8;
  bk[0] = frag_ld(kp0);        // keys 0..15, dh 0..31
  bk[1] = frag_ld(kp0 + 32);   // keys 0..15, dh 32..63
  bk[2] = frag_ld(kp1);        // keys 16..31, dh 0..31
  bk[3] = frag_ld(kp1 + 32);   // keys 16..31, dh 32..63
}

__global__ __launch_bounds__(512) void attn_kernel(const __bf16* __restrict__ Qb,
                                                   const __bf16* __restrict__ Kb,
                                                   const __bf16* __restrict__ Vt,
                                                   __bf16* __restrict__ Omean) {
  __shared__ float Pld[N_H][16 * 32];   // per-wave P transpose scratch (32 KB)
  __shared__ float Ored[16 * N_DH];     // head-mean accumulator (4 KB)

  int tid = threadIdx.x;
  int wave = tid >> 5, lane = tid & 31;
  int lgrp = lane & 15, half = lane >> 4;
  int b = blockIdx.x;
  int qt = blockIdx.y;
  int h = wave;

  for (int i = tid; i < 16 * N_DH; i += 512) Ored[i] = 0.f;
  __syncthreads();

  // Q fragments for this wave's 16 query rows (held in registers all loop long)
  const __bf16* qbase = Qb + ((size_t)(b * N_H + h) * N_S + qt * 16) * N_DH;
  v16bf aq0 = frag_ld(qbase + (size_t)lgrp * N_DH + half * 8);        // dh 0..31
  v16bf aq1 = frag_ld(qbase + (size_t)lgrp * N_DH + 32 + half * 8);   // dh 32..63

  const __bf16* kb = Kb + (size_t)(b * N_H + h) * N_S * N_DH;
  const __bf16* vb = Vt + (size_t)b * N_DH * N_S;

  float mrun[8], lrun[8];
  v8f acc[4] = {};
#pragma unroll
  for (int r = 0; r < 8; ++r) { mrun[r] = -1e30f; lrun[r] = 0.f; }

  float* myP = &Pld[wave][0];

  // one flash step: consume K frags `bkc` for key-tile kt, refill `bkn` with
  // the frags of key-tile kt+1 while the softmax VALU work proceeds.
  auto step = [&](int kt, v16bf (&bkc)[4], v16bf (&bkn)[4]) {
    int k0 = kt * 32;

    // prefetch two key-tiles ahead (global_prefetch_b8)
    int kp = ((kt + 2) & (N_S / 32 - 1)) * 32;
    __builtin_prefetch(kb + (size_t)(kp + lgrp) * N_DH, 0, 1);
    __builtin_prefetch(vb + (size_t)lgrp * N_S + kp, 0, 1);

    // --- scores: 16 queries x 32 keys (two 16x16 tiles, K-dim = 64 = 2 chunks)
    v8f c0 = {}, c1 = {};
    c0 = wmma_bf16(aq0, bkc[0], c0);
    c0 = wmma_bf16(aq1, bkc[1], c0);
    c1 = wmma_bf16(aq0, bkc[2], c1);
    c1 = wmma_bf16(aq1, bkc[3], c1);

    // issue V loads for THIS tile and K loads for the NEXT tile now, so they
    // are in flight while the softmax and LDS transpose proceed.
    v16bf bv[4];
#pragma unroll
    for (int t = 0; t < 4; ++t)
      bv[t] = frag_ld(vb + (size_t)(t * 16 + lgrp) * N_S + k0 + half * 8);
    load_k_frags(kb, ((kt + 1) & (N_S / 32 - 1)) * 32, lgrp, half, bkn);

    // --- online softmax (C layout: lane = column, VGPR r = row r+8*half)
#pragma unroll
    for (int r = 0; r < 8; ++r) {
      float mx = fmaxf(c0[r], c1[r]);
      mx = fmaxf(mx, __shfl_xor(mx, 1, 32));
      mx = fmaxf(mx, __shfl_xor(mx, 2, 32));
      mx = fmaxf(mx, __shfl_xor(mx, 4, 32));
      mx = fmaxf(mx, __shfl_xor(mx, 8, 32));
      float mnew = fmaxf(mrun[r], mx);
      float fr = __expf(mrun[r] - mnew);
      mrun[r] = mnew;
      lrun[r] *= fr;
#pragma unroll
      for (int t = 0; t < 4; ++t) acc[t][r] *= fr;
      float e0 = __expf(c0[r] - mnew);
      float e1 = __expf(c1[r] - mnew);
      c0[r] = e0; c1[r] = e1;
      float sm = e0 + e1;
      sm += __shfl_xor(sm, 1, 32);
      sm += __shfl_xor(sm, 2, 32);
      sm += __shfl_xor(sm, 4, 32);
      sm += __shfl_xor(sm, 8, 32);
      lrun[r] += sm;
    }

    // --- transpose P through LDS: C-layout store, A-layout reload
#pragma unroll
    for (int r = 0; r < 8; ++r) {
      myP[(r + 8 * half) * 32 + lgrp]      = c0[r];
      myP[(r + 8 * half) * 32 + 16 + lgrp] = c1[r];
    }
    // same-wave DS ops are in order: reload is safe
    v4f p0a = *(const v4f*)&myP[lgrp * 32 + half * 8];
    v4f p0b = *(const v4f*)&myP[lgrp * 32 + half * 8 + 4];
    v4f p1a = *(const v4f*)&myP[lgrp * 32 + 16 + half * 8];
    v4f p1b = *(const v4f*)&myP[lgrp * 32 + 16 + half * 8 + 4];
    v16bf ap;
#pragma unroll
    for (int i = 0; i < 4; ++i) {
      ap[i]      = f2bf(p0a[i]);
      ap[4 + i]  = f2bf(p0b[i]);
      ap[8 + i]  = f2bf(p1a[i]);
      ap[12 + i] = f2bf(p1b[i]);
    }

    // --- P @ V : V transposed so each lane reads contiguous key runs
#pragma unroll
    for (int t = 0; t < 4; ++t)
      acc[t] = wmma_bf16(ap, bv[t], acc[t]);
  };

  // ping-pong K buffers; loop unrolled by 2 so no buffer copies are needed
  v16bf bkA[4], bkB[4];
  load_k_frags(kb, 0, lgrp, half, bkA);
  for (int kt = 0; kt < N_S / 32; kt += 2) {
    step(kt,     bkA, bkB);
    step(kt + 1, bkB, bkA);
  }

  // normalize and reduce mean over heads
#pragma unroll
  for (int r = 0; r < 8; ++r) {
    float inv = 1.0f / lrun[r];
#pragma unroll
    for (int t = 0; t < 4; ++t) acc[t][r] *= inv;
  }
#pragma unroll
  for (int t = 0; t < 4; ++t)
#pragma unroll
    for (int r = 0; r < 8; ++r)
      atomicAdd(&Ored[(r + 8 * half) * N_DH + t * 16 + lgrp], acc[t][r]);
  __syncthreads();

  __bf16* op = Omean + ((size_t)b * N_S + qt * 16) * N_DH;
  for (int i = tid; i < 16 * N_DH; i += 512)
    op[i] = f2bf(Ored[i] * (1.0f / N_H));
}

// ---- output projection: out[s,:] = Omean[s,:] @ Wo + bo  (M=8192,N=1024,K=64)
__global__ __launch_bounds__(256) void outproj_kernel(const __bf16* __restrict__ Omean,
                                                      const __bf16* __restrict__ Wot,
                                                      const float* __restrict__ bo,
                                                      float* __restrict__ out) {
  int wave = threadIdx.x >> 5;
  int lane = threadIdx.x & 31;
  int lgrp = lane & 15, half = lane >> 4;
  int rbase = (blockIdx.x * 8 + wave) * 16;
  int ng = blockIdx.y;                       // 64-wide column group of 1024

  const __bf16* ar = Omean + (size_t)(rbase + lgrp) * N_DH + half * 8;
  v16bf a0 = frag_ld(ar);        // k 0..31
  v16bf a1 = frag_ld(ar + 32);   // k 32..63

#pragma unroll
  for (int t = 0; t < 4; ++t) {
    int n = ng * 64 + t * 16 + lgrp;
    const __bf16* wr = Wot + (size_t)n * N_DH + half * 8;
    v16bf b0 = frag_ld(wr);
    v16bf b1 = frag_ld(wr + 32);
    v8f c = {};
    c = wmma_bf16(a0, b0, c);
    c = wmma_bf16(a1, b1, c);
    float bb = bo[n];
#pragma unroll
    for (int r = 0; r < 8; ++r)
      out[(size_t)(rbase + r + 8 * half) * N_D + n] = c[r] + bb;
  }
}

// ---------------------------------------------------------------------------

extern "C" void kernel_launch(void* const* d_in, const int* in_sizes, int n_in,
                              void* d_out, int out_size, void* d_ws, size_t ws_size,
                              hipStream_t stream) {
  const float* query = (const float*)d_in[0];   // [B,S,1024]
  const float* key   = (const float*)d_in[1];
  const float* value = (const float*)d_in[2];
  const float* Wq    = (const float*)d_in[3];   // [H,1024,64]
  const float* bq    = (const float*)d_in[4];   // [H,64]
  const float* Wk    = (const float*)d_in[5];
  const float* bk    = (const float*)d_in[6];
  const float* Wv    = (const float*)d_in[7];   // [1024,64]
  const float* bv    = (const float*)d_in[8];   // [64]
  const float* Wo    = (const float*)d_in[9];   // [64,1024]
  const float* bo    = (const float*)d_in[10];  // [1024]
  float* out = (float*)d_out;                   // [B,S,1024] fp32

  // workspace layout (bf16 staging, ~91 MB, fits in 192 MB L2)
  char* ws = (char*)d_ws;
  size_t off = 0;
  auto alloc = [&](size_t bytes) { char* p = ws + off; off = (off + bytes + 255) & ~(size_t)255; return p; };
  __bf16* queryb = (__bf16*)alloc((size_t)BS * N_D * 2);
  __bf16* keyb   = (__bf16*)alloc((size_t)BS * N_D * 2);
  __bf16* valueb = (__bf16*)alloc((size_t)BS * N_D * 2);
  __bf16* Wqt    = (__bf16*)alloc((size_t)N_H * N_DH * N_D * 2);
  __bf16* Wkt    = (__bf16*)alloc((size_t)N_H * N_DH * N_D * 2);
  __bf16* Wvt    = (__bf16*)alloc((size_t)N_DH * N_D * 2);
  __bf16* Wot    = (__bf16*)alloc((size_t)N_D * N_DH * 2);
  __bf16* Qb     = (__bf16*)alloc((size_t)N_B * N_H * N_S * N_DH * 2);
  __bf16* Kb     = (__bf16*)alloc((size_t)N_B * N_H * N_S * N_DH * 2);
  __bf16* Vt     = (__bf16*)alloc((size_t)N_B * N_DH * N_S * 2);
  __bf16* Omean  = (__bf16*)alloc((size_t)BS * N_DH * 2);

  const long nAct = (long)BS * N_D;
  cvt_bf16_kernel<<<dim3((nAct + 255) / 256), dim3(256), 0, stream>>>(query, queryb, nAct);
  cvt_bf16_kernel<<<dim3((nAct + 255) / 256), dim3(256), 0, stream>>>(key,   keyb,   nAct);
  cvt_bf16_kernel<<<dim3((nAct + 255) / 256), dim3(256), 0, stream>>>(value, valueb, nAct);

  const long nW = (long)N_H * N_D * N_DH;  // 1,048,576
  btrans_bf16_kernel<<<dim3((nW + 255) / 256), dim3(256), 0, stream>>>(Wq, Wqt, N_D, N_DH, nW);
  btrans_bf16_kernel<<<dim3((nW + 255) / 256), dim3(256), 0, stream>>>(Wk, Wkt, N_D, N_DH, nW);
  const long nWv = (long)N_D * N_DH;
  btrans_bf16_kernel<<<dim3((nWv + 255) / 256), dim3(256), 0, stream>>>(Wv, Wvt, N_D, N_DH, nWv);
  btrans_bf16_kernel<<<dim3((nWv + 255) / 256), dim3(256), 0, stream>>>(Wo, Wot, N_DH, N_D, nWv);

  // per-head Q/K projections; Q pre-scaled by d_head^-0.5 = 0.125
  dim3 gproj(BS / (16 * 8), N_H);
  proj_qk_kernel<<<gproj, dim3(256), 0, stream>>>(queryb, Wqt, bq, Qb, 0.125f);
  proj_qk_kernel<<<gproj, dim3(256), 0, stream>>>(keyb,   Wkt, bk, Kb, 1.0f);
  proj_v_kernel<<<dim3(BS / (16 * 8)), dim3(256), 0, stream>>>(valueb, Wvt, bv, Vt);

  // flash attention + head mean: one block per (batch, 16-query tile)
  attn_kernel<<<dim3(N_B, N_S / 16), dim3(512), 0, stream>>>(Qb, Kb, Vt, Omean);

  // output projection -> fp32 d_out
  outproj_kernel<<<dim3(BS / (16 * 8), N_D / 64), dim3(256), 0, stream>>>(Omean, Wot, bo, out);
}
